// EncoderLayer_3152505995303
// MI455X (gfx1250) — compile-verified, matrix operand order checked by reading, not run on hardware
//
#include <hip/hip_runtime.h>

// ---------------------------------------------------------------------------
// Transformer encoder layer for MI455X (gfx1250): bf16 WMMA everywhere,
// async-to-LDS DMA for the transpose/staging kernels.
// B=2, S=2048, D=1024, H=16, DK=64, DFF=4096, M=B*S=4096.
// ---------------------------------------------------------------------------

#define DEV __device__ __forceinline__

typedef __attribute__((ext_vector_type(16))) __bf16 v16bf;
typedef __attribute__((ext_vector_type(8)))  __bf16 v8bf;
typedef __attribute__((ext_vector_type(8)))  float  v8f;

union Frag16 { v16bf v; v8bf h[2]; __bf16 e[16]; };

DEV __bf16 f2b(float f) {
  unsigned u = __builtin_bit_cast(unsigned, f);
  unsigned r = u + 0x7FFFu + ((u >> 16) & 1u);   // round-to-nearest-even
  unsigned short s = (unsigned short)(r >> 16);
  return __builtin_bit_cast(__bf16, s);
}

// --- CDNA5 async global->LDS DMA (ASYNCcnt-tracked), inline-asm portable ---
DEV void async_lds_b64(void* lds, const void* g) {
  unsigned l = (unsigned)(uintptr_t)lds;            // flat[31:0] == LDS addr
  unsigned long long a = (unsigned long long)(uintptr_t)g;
  asm volatile("global_load_async_to_lds_b64 %0, %1, off"
               :: "v"(l), "v"(a) : "memory");
}
DEV void async_lds_b128(void* lds, const void* g) {
  unsigned l = (unsigned)(uintptr_t)lds;
  unsigned long long a = (unsigned long long)(uintptr_t)g;
  asm volatile("global_load_async_to_lds_b128 %0, %1, off"
               :: "v"(l), "v"(a) : "memory");
}
DEV void wait_async0() {
  asm volatile("s_wait_asynccnt 0x0" ::: "memory");
}

// Load a 16x32 bf16 WMMA fragment from "leading-dim-major" storage.
// A-matrix:  row0 = M tile base,  ld = K stride       (rows are M)
// B-matrix:  row0 = N tile base,  ld = K stride       (rows are N, i.e. W^T)
// Per ISA 16-bit layout: lane l<16 holds row l, K = {hi*8..+7, 16+hi*8..+7}.
DEV v16bf frag_ld(const __bf16* __restrict__ base, int ld, int row0, int k0,
                  int l15, int hi) {
  const __bf16* p = base + (size_t)(row0 + l15) * (size_t)ld + (k0 + hi * 8);
  Frag16 f;
  f.h[0] = *(const v8bf*)(p);
  f.h[1] = *(const v8bf*)(p + 16);
  return f.v;
}

DEV v8f wmma_bf16(v16bf a, v16bf b, v8f c) {
  // (neg_a, A, neg_b, B, c_mod, C, reuse_a, reuse_b)
  return __builtin_amdgcn_wmma_f32_16x16x32_bf16(false, a, false, b,
                                                 (short)0, c, false, false);
}

// ---------------------------------------------------------------------------
// LayerNorm (torch-style: unbiased var (ddof=1), eps added to std) -> bf16.
// One 256-thread block per row of D=1024 (each thread owns a float4).
// ---------------------------------------------------------------------------
__global__ __launch_bounds__(256) void ln_kernel(
    const float* __restrict__ X, const float* __restrict__ g,
    const float* __restrict__ be, __bf16* __restrict__ Hout, int Dm) {
  const int row = blockIdx.x;
  const int tid = threadIdx.x;
  const float* x = X + (size_t)row * Dm;
  float4 v = ((const float4*)x)[tid];
  float s  = v.x + v.y + v.z + v.w;
  float ss = v.x * v.x + v.y * v.y + v.z * v.z + v.w * v.w;
  for (int o = 16; o > 0; o >>= 1) {
    s  += __shfl_xor(s, o, 32);
    ss += __shfl_xor(ss, o, 32);
  }
  __shared__ float rs[8], rss[8], tot[2];
  if ((tid & 31) == 0) { rs[tid >> 5] = s; rss[tid >> 5] = ss; }
  __syncthreads();
  if (tid == 0) {
    float a = 0.f, b2 = 0.f;
    for (int w = 0; w < 8; ++w) { a += rs[w]; b2 += rss[w]; }
    tot[0] = a; tot[1] = b2;
  }
  __syncthreads();
  const float mean = tot[0] / (float)Dm;
  const float var  = (tot[1] - tot[0] * tot[0] / (float)Dm) / (float)(Dm - 1);
  const float inv  = 1.0f / (sqrtf(var) + 1e-6f);
  const int c = tid * 4;
  float e[4] = { v.x, v.y, v.z, v.w };
  for (int i = 0; i < 4; ++i)
    Hout[(size_t)row * Dm + c + i] = f2b(g[c + i] * ((e[i] - mean) * inv) + be[c + i]);
}

// ---------------------------------------------------------------------------
// Weight transpose + cast: W[K,N] f32 -> Wt[N,K] bf16 (32x32 LDS tiles).
// Tile rows padded to 36 floats so every async b128 LDS dest is 16B-aligned.
// ---------------------------------------------------------------------------
__global__ __launch_bounds__(256) void wcast_kernel(
    const float* __restrict__ W, __bf16* __restrict__ Wt, int K, int N) {
  __shared__ float tile[32][36];
  const int n0 = blockIdx.x * 32, k0 = blockIdx.y * 32;
  const int tid = threadIdx.x;
  const int tx = tid & 31, ty = tid >> 5;
  {   // async DMA: 256 threads x 16B = full 32x32 f32 tile
    const int i = tid >> 3, c = (tid & 7) * 4;
    async_lds_b128(&tile[i][c], &W[(size_t)(k0 + i) * N + (n0 + c)]);
  }
  wait_async0();
  __syncthreads();
  for (int i = ty; i < 32; i += 8)
    Wt[(size_t)(n0 + i) * K + (k0 + tx)] = f2b(tile[tx][i]);
}

// V[b*S+s, D] bf16 -> Vt[b*D + c, S] bf16 (per-batch transpose).
// Tile rows padded to 36 bf16 so async b64 LDS dests are 8B-aligned.
__global__ __launch_bounds__(256) void vtrans_kernel(
    const __bf16* __restrict__ V, __bf16* __restrict__ Vt, int S_, int Dm) {
  __shared__ __bf16 tile[32][36];
  const int s0 = blockIdx.x * 32, c0 = blockIdx.y * 32, b = blockIdx.z;
  const int tid = threadIdx.x;
  const int tx = tid & 31, ty = tid >> 5;
  if (tid < 128) {   // async DMA: 128 threads x 8B = 1 KB, half of the 32x32 bf16 tile
    const int i = tid >> 2, c = (tid & 3) * 8;
    async_lds_b64(&tile[i][c], &V[((size_t)b * S_ + s0 + i) * Dm + (c0 + c)]);
  }
  wait_async0();
  __syncthreads();
  for (int i = ty; i < 32; i += 8)
    Vt[((size_t)b * Dm + c0 + i) * S_ + (s0 + tx)] = tile[tx][i];
}

// ---------------------------------------------------------------------------
// GEMM: C[M,N] = act(A[M,K] @ Wt[N,K]^T + bias) (+ resid). bf16 in, f32 acc.
// 256 threads = 8 waves; wave tile 32x64 (2x4 wmma accs); block tile 64x256.
// ---------------------------------------------------------------------------
template <bool RELU, bool RESID, bool OBF>
__global__ __launch_bounds__(256) void gemm_bf16(
    const __bf16* __restrict__ A, const __bf16* __restrict__ Wt,
    const float* __restrict__ bias, const float* __restrict__ resid,
    float* __restrict__ Of, __bf16* __restrict__ Ob, int M, int N, int K) {
  const int tid  = threadIdx.x;
  const int wave = tid >> 5, lane = tid & 31;
  const int l15  = lane & 15, hi = lane >> 4;
  const int m0 = blockIdx.y * 64 + (wave >> 2) * 32;
  const int n0 = blockIdx.x * 256 + (wave & 3) * 64;

  v8f acc[2][4];
  for (int i = 0; i < 2; ++i)
    for (int j = 0; j < 4; ++j) acc[i][j] = (v8f){};

  for (int k0 = 0; k0 < K; k0 += 32) {
    if (k0 + 128 < K) {
      __builtin_prefetch(A  + (size_t)(m0 + l15) * K + k0 + 128, 0, 0);
      __builtin_prefetch(Wt + (size_t)(n0 + l15) * K + k0 + 128, 0, 0);
    }
    v16bf a0 = frag_ld(A, K, m0,      k0, l15, hi);
    v16bf a1 = frag_ld(A, K, m0 + 16, k0, l15, hi);
    v16bf b0 = frag_ld(Wt, K, n0,      k0, l15, hi);
    v16bf b1 = frag_ld(Wt, K, n0 + 16, k0, l15, hi);
    v16bf b2 = frag_ld(Wt, K, n0 + 32, k0, l15, hi);
    v16bf b3 = frag_ld(Wt, K, n0 + 48, k0, l15, hi);
    acc[0][0] = wmma_bf16(a0, b0, acc[0][0]);
    acc[0][1] = wmma_bf16(a0, b1, acc[0][1]);
    acc[0][2] = wmma_bf16(a0, b2, acc[0][2]);
    acc[0][3] = wmma_bf16(a0, b3, acc[0][3]);
    acc[1][0] = wmma_bf16(a1, b0, acc[1][0]);
    acc[1][1] = wmma_bf16(a1, b1, acc[1][1]);
    acc[1][2] = wmma_bf16(a1, b2, acc[1][2]);
    acc[1][3] = wmma_bf16(a1, b3, acc[1][3]);
  }

  // Epilogue. C layout: VGPR r -> row (hi*8 + r), col l15 within each 16x16.
  for (int i = 0; i < 2; ++i) {
    for (int j = 0; j < 4; ++j) {
      const int n = n0 + j * 16 + l15;
      const float bv = bias[n];
      for (int r = 0; r < 8; ++r) {
        const int m = m0 + i * 16 + hi * 8 + r;
        float val = acc[i][j][r] + bv;
        if (RESID) val += resid[(size_t)m * N + n];
        if (RELU)  val = fmaxf(val, 0.0f);
        if (OBF) Ob[(size_t)m * N + n] = f2b(val);
        else     Of[(size_t)m * N + n] = val;
      }
    }
  }
}

// ---------------------------------------------------------------------------
// Attention: one block per (16-query strip, head, batch). 8 waves, each wave
// owns a 256-key span. Raw scores -> LDS (f32, 16 x S); mask applied during
// the softmax pass with contiguous b128 mask-byte loads; P@V with the
// pre-transposed Vt[b*D + h*64 + dk, S]; cross-wave reduce via LDS.
// Dynamic LDS: 16*S*4 + 16*16*4 + 16*4 = 132,160 B (< 160 KB CU-mode cap).
// ---------------------------------------------------------------------------
__global__ __launch_bounds__(256) void attn_kernel(
    const __bf16* __restrict__ Q, const __bf16* __restrict__ Kb,
    const __bf16* __restrict__ Vt, const unsigned char* __restrict__ mask,
    __bf16* __restrict__ Ctx, int Bn, int S_, int Hn, int Dm) {
  extern __shared__ char smem[];
  float* sc   = (float*)smem;                          // [16][S] scores/probs
  float* red  = (float*)(smem + (size_t)16 * S_ * 4);  // [16][16]
  float* rden = red + 256;                             // [16]

  const int tid  = threadIdx.x;
  const int wave = tid >> 5, lane = tid & 31;
  const int l15  = lane & 15, hi = lane >> 4;
  const int q0 = blockIdx.x * 16;
  const int h  = blockIdx.y;
  const int b  = blockIdx.z;
  const int rowbase = b * S_;          // row offset into [B*S, D]
  const int col = h * 64;              // head column in D

  // Q fragments held in registers for the whole kernel (16 x 64, two k-steps).
  v16bf aq0 = frag_ld(Q, Dm, rowbase + q0, col,      l15, hi);
  v16bf aq1 = frag_ld(Q, Dm, rowbase + q0, col + 32, l15, hi);

  // ---- raw scores = (Q K^T) / sqrt(64) into LDS ----------------------------
  const float scal = 0.125f;
  for (int t = 0; t < 16; ++t) {
    const int kk = wave * 256 + t * 16;       // 16-key tile
    v16bf bk0 = frag_ld(Kb, Dm, rowbase + kk, col,      l15, hi);
    v16bf bk1 = frag_ld(Kb, Dm, rowbase + kk, col + 32, l15, hi);
    v8f s = (v8f){};
    s = wmma_bf16(aq0, bk0, s);
    s = wmma_bf16(aq1, bk1, s);
    const int k = kk + l15;
    for (int r = 0; r < 8; ++r)
      sc[(hi * 8 + r) * S_ + k] = s[r] * scal;
  }
  __syncthreads();

  // ---- mask + softmax over full rows (16 threads/row, 128 keys each) -------
  {
    const int row = tid >> 4, seg = tid & 15;
    float* p = sc + row * S_ + seg * 128;
    const uint4* mv =
        (const uint4*)(mask + ((size_t)b * S_ + q0 + row) * (size_t)S_ +
                       (size_t)seg * 128);
    float mx = -3.4e38f;
    for (int c = 0; c < 8; ++c) {                 // 8 x 16 mask bytes (b128)
      union { uint4 u; unsigned char bby[16]; } mu;
      mu.u = mv[c];
      for (int j = 0; j < 16; ++j) {
        float v = p[c * 16 + j];
        if (mu.bby[j]) v = -1e9f;
        p[c * 16 + j] = v;                        // masked score in place
        mx = fmaxf(mx, v);
      }
    }
    red[row * 16 + seg] = mx;
    __syncthreads();
    if (seg == 0) {
      float m = red[row * 16];
      for (int i = 1; i < 16; ++i) m = fmaxf(m, red[row * 16 + i]);
      red[row * 16] = m;
    }
    __syncthreads();
    const float m = red[row * 16];
    float sum = 0.f;
    for (int i = 0; i < 128; ++i) {
      float e = __expf(p[i] - m);
      p[i] = e;                       // probs in place (unique owner per elem)
      sum += e;
    }
    __syncthreads();
    red[row * 16 + seg] = sum;
    __syncthreads();
    if (tid < 16) {
      float t = 0.f;
      for (int i = 0; i < 16; ++i) t += red[tid * 16 + i];
      rden[tid] = 1.0f / t;
    }
    __syncthreads();
  }

  // ---- ctx partial = P @ V over this wave's 256-key span -------------------
  v8f cv[4];
  for (int j = 0; j < 4; ++j) cv[j] = (v8f){};
  const int vrow = b * Dm + col;       // Vt row base for this (b, h)
  for (int t2 = 0; t2 < 8; ++t2) {
    const int kk = wave * 256 + t2 * 32;    // 32-key k-step
    // A fragment: probs row l15, keys kk+hi*8..(+7) and +16..(+23), f32->bf16
    Frag16 fa;
    const float* pa = sc + l15 * S_ + kk + hi * 8;
    for (int i = 0; i < 8; ++i) fa.e[i]     = f2b(pa[i]);
    for (int i = 0; i < 8; ++i) fa.e[8 + i] = f2b(pa[16 + i]);
    for (int j = 0; j < 4; ++j) {
      v16bf bv = frag_ld(Vt, S_, vrow + j * 16, kk, l15, hi);
      cv[j] = wmma_bf16(fa.v, bv, cv[j]);
    }
  }
  __syncthreads();     // all P reads from sc done; safe to overlay ctxred

  // ---- cross-wave reduce + normalize + store -------------------------------
  float* ctxred = sc;                  // [8][16][64] overlay (32 KB)
  for (int j = 0; j < 4; ++j)
    for (int r = 0; r < 8; ++r) {
      const int q = hi * 8 + r, dk = j * 16 + l15;
      ctxred[(wave * 16 + q) * 64 + dk] = cv[j][r];
    }
  __syncthreads();
  for (int e0 = tid; e0 < 16 * 64; e0 += 256) {
    const int q = e0 >> 6, dk = e0 & 63;
    float s = 0.f;
    for (int w = 0; w < 8; ++w) s += ctxred[(w * 16 + q) * 64 + dk];
    s *= rden[q];
    Ctx[((size_t)rowbase + q0 + q) * Dm + col + dk] = f2b(s);
  }
}

// ---------------------------------------------------------------------------
// Host orchestration.
// ---------------------------------------------------------------------------
extern "C" void kernel_launch(void* const* d_in, const int* in_sizes, int n_in,
                              void* d_out, int out_size, void* d_ws, size_t ws_size,
                              hipStream_t stream) {
  (void)in_sizes; (void)n_in; (void)out_size; (void)ws_size;
  constexpr int Bc = 2, Sc = 2048, Dc = 1024, Hc = 16, Fc = 4096;
  constexpr int Mc = Bc * Sc;                         // 4096 rows

  const float* x   = (const float*)d_in[0];
  const unsigned char* mask = (const unsigned char*)d_in[1];
  const float* wq  = (const float*)d_in[2];
  const float* bq  = (const float*)d_in[3];
  const float* wk  = (const float*)d_in[4];
  const float* bk  = (const float*)d_in[5];
  const float* wv  = (const float*)d_in[6];
  const float* bv  = (const float*)d_in[7];
  const float* wo  = (const float*)d_in[8];
  const float* bo  = (const float*)d_in[9];
  const float* w1  = (const float*)d_in[10];
  const float* b1  = (const float*)d_in[11];
  const float* w2  = (const float*)d_in[12];
  const float* b2  = (const float*)d_in[13];
  const float* g1  = (const float*)d_in[14];
  const float* be1 = (const float*)d_in[15];
  const float* g2  = (const float*)d_in[16];
  const float* be2 = (const float*)d_in[17];
  float* out = (float*)d_out;

  char* ws = (char*)d_ws;
  size_t off = 0;
  auto take = [&](size_t bytes) -> char* {
    char* p = ws + off;
    off += (bytes + 255) & ~(size_t)255;
    return p;
  };
  __bf16* h1  = (__bf16*)take((size_t)Mc * Dc * 2);
  __bf16* qb  = (__bf16*)take((size_t)Mc * Dc * 2);
  __bf16* kb  = (__bf16*)take((size_t)Mc * Dc * 2);
  __bf16* vb  = (__bf16*)take((size_t)Mc * Dc * 2);
  __bf16* vt  = (__bf16*)take((size_t)Bc * Dc * Sc * 2);
  __bf16* ctx = (__bf16*)take((size_t)Mc * Dc * 2);
  float*  x1  = (float*) take((size_t)Mc * Dc * 4);
  __bf16* h2  = (__bf16*)take((size_t)Mc * Dc * 2);
  __bf16* ff1 = (__bf16*)take((size_t)Mc * Fc * 2);
  __bf16* wqt = (__bf16*)take((size_t)Dc * Dc * 2);
  __bf16* wkt = (__bf16*)take((size_t)Dc * Dc * 2);
  __bf16* wvt = (__bf16*)take((size_t)Dc * Dc * 2);
  __bf16* wot = (__bf16*)take((size_t)Dc * Dc * 2);
  __bf16* w1t = (__bf16*)take((size_t)Fc * Dc * 2);
  __bf16* w2t = (__bf16*)take((size_t)Dc * Fc * 2);

  const dim3 blk(256);

  // Weight transposes (f32 -> bf16, [K,N] -> [N,K]), async-to-LDS staged.
  wcast_kernel<<<dim3(Dc / 32, Dc / 32), blk, 0, stream>>>(wq, wqt, Dc, Dc);
  wcast_kernel<<<dim3(Dc / 32, Dc / 32), blk, 0, stream>>>(wk, wkt, Dc, Dc);
  wcast_kernel<<<dim3(Dc / 32, Dc / 32), blk, 0, stream>>>(wv, wvt, Dc, Dc);
  wcast_kernel<<<dim3(Dc / 32, Dc / 32), blk, 0, stream>>>(wo, wot, Dc, Dc);
  wcast_kernel<<<dim3(Fc / 32, Dc / 32), blk, 0, stream>>>(w1, w1t, Dc, Fc);
  wcast_kernel<<<dim3(Dc / 32, Fc / 32), blk, 0, stream>>>(w2, w2t, Fc, Dc);

  // Sublayer 1: pre-norm MHA + residual.
  ln_kernel<<<Mc, blk, 0, stream>>>(x, g1, be1, h1, Dc);
  gemm_bf16<false, false, true><<<dim3(Dc / 256, Mc / 64), blk, 0, stream>>>(
      h1, wqt, bq, nullptr, nullptr, qb, Mc, Dc, Dc);
  gemm_bf16<false, false, true><<<dim3(Dc / 256, Mc / 64), blk, 0, stream>>>(
      h1, wkt, bk, nullptr, nullptr, kb, Mc, Dc, Dc);
  gemm_bf16<false, false, true><<<dim3(Dc / 256, Mc / 64), blk, 0, stream>>>(
      h1, wvt, bv, nullptr, nullptr, vb, Mc, Dc, Dc);
  vtrans_kernel<<<dim3(Sc / 32, Dc / 32, Bc), blk, 0, stream>>>(vb, vt, Sc, Dc);

  const size_t attn_lds = (size_t)16 * Sc * 4 + 16 * 16 * 4 + 16 * 4;
  attn_kernel<<<dim3(Sc / 16, Hc, Bc), blk, attn_lds, stream>>>(
      qb, kb, vt, mask, ctx, Bc, Sc, Hc, Dc);

  gemm_bf16<false, true, false><<<dim3(Dc / 256, Mc / 64), blk, 0, stream>>>(
      ctx, wot, bo, x, x1, nullptr, Mc, Dc, Dc);

  // Sublayer 2: pre-norm FFN + residual.
  ln_kernel<<<Mc, blk, 0, stream>>>(x1, g2, be2, h2, Dc);
  gemm_bf16<true, false, true><<<dim3(Fc / 256, Mc / 64), blk, 0, stream>>>(
      h2, w1t, b1, nullptr, nullptr, ff1, Mc, Fc, Dc);
  gemm_bf16<false, true, false><<<dim3(Dc / 256, Mc / 64), blk, 0, stream>>>(
      ff1, w2t, b2, x1, out, nullptr, Mc, Dc, Fc);
}
